// AttnBlock_NLP_87230785782270
// MI455X (gfx1250) — matrix operand order repaired and checked by reading, not compiled, not run on hardware
//
#include <hip/hip_runtime.h>
#include <hip/hip_bf16.h>

// ---------------------------------------------------------------------------
// Fused AttnBlock for MI455X (gfx1250, wave32, WMMA).
//   out = xn + ( softmax_rows(outer(k,q)) @ v / sqrt(R) ) @ r_w^T + r_b
// One block = 16 samples, 128 threads (4 waves). Everything stays in LDS.
// GEMMs via v_wmma_f32_16x16x32_bf16 (bf16 A/B, fp32 accumulate).
// Weights are pre-converted to bf16 and pre-swizzled into WMMA B-fragment
// order in d_ws by a tiny prep kernel (one aligned 32B load per lane per
// fragment, no cvt in the hot loops).
// ---------------------------------------------------------------------------

#define Bdim   16384
#define Cdim   512
#define Rdim   128
#define Gnum   32
#define GS     16          // channels per group
#define SPB    16          // samples per block
#define NTHR   128         // 4 wave32

typedef __attribute__((ext_vector_type(16))) __bf16 v16bf;
typedef __attribute__((ext_vector_type(8)))  __bf16 v8bf;
typedef __attribute__((ext_vector_type(8)))  float  v8f;

// LDS row paddings (keep 16B alignment of 8-elem bf16 chunks, skew banks)
#define XNB_STRIDE 520     // bf16, 1040 B / row
#define KQV_STRIDE 132     // f32 (k only)
#define QV_STRIDE  264     // f32, interleaved (q,v) pairs
#define ATT_STRIDE 136     // bf16, 272 B / row

// Swizzled-weight workspace layout (bf16):
//   [0*65536)  k_w fragments   : ((nt*16 + kt)*32 + lane)*16 + j   (nt<8, kt<16)
//   [1*65536)  q_w fragments
//   [2*65536)  v_w fragments
//   [3*65536)  r_w fragments   : ((nt*4  + kt)*32 + lane)*16 + j   (nt<32, kt<4)
#define WS_MAT_ELEMS 65536
#define WS_TOTAL_ELEMS (4 * WS_MAT_ELEMS)

__device__ __forceinline__ v8f wmma_bf16(v16bf a, v16bf b, v8f c) {
  // (neg_a, A, neg_b, B, c_mod, C, reuse_a, reuse_b)
  return __builtin_amdgcn_wmma_f32_16x16x32_bf16(false, a, false, b,
                                                 (short)0, c, false, false);
}

// Fallback path: load 16 contiguous fp32 (64B) and convert to bf16x16.
__device__ __forceinline__ v16bf load_bfrag_f32(const float* p) {
  const float4* q4 = (const float4*)p;
  float4 a0 = q4[0], a1 = q4[1], a2 = q4[2], a3 = q4[3];
  v16bf b;
  b[0]  = (__bf16)a0.x; b[1]  = (__bf16)a0.y; b[2]  = (__bf16)a0.z; b[3]  = (__bf16)a0.w;
  b[4]  = (__bf16)a1.x; b[5]  = (__bf16)a1.y; b[6]  = (__bf16)a1.z; b[7]  = (__bf16)a1.w;
  b[8]  = (__bf16)a2.x; b[9]  = (__bf16)a2.y; b[10] = (__bf16)a2.z; b[11] = (__bf16)a2.w;
  b[12] = (__bf16)a3.x; b[13] = (__bf16)a3.y; b[14] = (__bf16)a3.z; b[15] = (__bf16)a3.w;
  return b;
}

// A fragment (16x32 bf16, lane m=l&15, h=l>>4;
// elems 0..15 -> K = 16h + j, contiguous) from an LDS row.
__device__ __forceinline__ v16bf load_afrag_lds(const __bf16* row, int kb, int h) {
  v8bf lo = *(const v8bf*)(row + kb + 16 * h);
  v8bf hi = *(const v8bf*)(row + kb + 16 * h + 8);
  return __builtin_shufflevector(lo, hi, 0,1,2,3,4,5,6,7,8,9,10,11,12,13,14,15);
}

// ---------------------------------------------------------------------------
// Prep kernel: convert fp32 weights -> bf16, pre-swizzled into per-fragment
// lane-major order so the GEMM loops do one contiguous 32B load per lane.
// ---------------------------------------------------------------------------
__global__ void prep_weights(const float* __restrict__ kw,
                             const float* __restrict__ qw,
                             const float* __restrict__ vw,
                             const float* __restrict__ rw,
                             unsigned short* __restrict__ ws_us) {
  __bf16* ws = (__bf16*)ws_us;
  const int idx = blockIdx.x * blockDim.x + threadIdx.x;   // < WS_TOTAL_ELEMS
  const int mat = idx >> 16;            // 0=k, 1=q, 2=v, 3=r
  const int e   = idx & (WS_MAT_ELEMS - 1);
  const int j    = e & 15;
  const int lane = (e >> 4) & 31;
  const int n = lane & 15, h = lane >> 4;
  float val;
  if (mat < 3) {
    const int kt = (e >> 9) & 15;
    const int nt = e >> 13;
    const int r = nt * 16 + n;                 // output row of W  (N)
    const int c = kt * 32 + 16 * h + j;        // input channel    (K)
    const float* w = (mat == 0) ? kw : (mat == 1) ? qw : vw;
    val = w[(size_t)r * Cdim + c];
  } else {
    const int kt = (e >> 9) & 3;
    const int nt = e >> 11;
    const int c = nt * 16 + n;                 // output channel   (N)
    const int r = kt * 32 + 16 * h + j;        // reduced channel  (K)
    val = rw[(size_t)c * Rdim + r];
  }
  ws[idx] = (__bf16)val;
}

// ---------------------------------------------------------------------------
// Main fused kernel. USE_BF: B fragments come pre-swizzled bf16 from d_ws.
// ---------------------------------------------------------------------------
template <bool USE_BF>
__global__ __launch_bounds__(NTHR, 2)
void attn_block_fused(const float* __restrict__ x,
                      const float* __restrict__ gamma,
                      const float* __restrict__ beta,
                      const float* __restrict__ kw, const float* __restrict__ kbias,
                      const float* __restrict__ qw, const float* __restrict__ qbias,
                      const float* __restrict__ vw, const float* __restrict__ vbias,
                      const float* __restrict__ rw, const float* __restrict__ rbias,
                      const unsigned short* __restrict__ ws_us,
                      float* __restrict__ out) {
  __shared__ __align__(16) __bf16 s_xnb[SPB * XNB_STRIDE];   // normalized x, bf16
  __shared__ __align__(16) float  s_k[SPB * KQV_STRIDE];
  __shared__ __align__(16) float  s_qv[SPB * QV_STRIDE];     // interleaved (q,v)
  __shared__ __align__(16) __bf16 s_attn[SPB * ATT_STRIDE];  // attn out, bf16
  __shared__ float s_mu[SPB][Gnum];
  __shared__ float s_rstd[SPB][Gnum];
  __shared__ float s_qmax[SPB];
  __shared__ float s_qmin[SPB];

  const __bf16* wsb = (const __bf16*)ws_us;
  const __bf16* kfr = wsb;
  const __bf16* qfr = wsb + 1 * WS_MAT_ELEMS;
  const __bf16* vfr = wsb + 2 * WS_MAT_ELEMS;
  const __bf16* rfr = wsb + 3 * WS_MAT_ELEMS;

  const int tid  = threadIdx.x;
  const int wave = tid >> 5;
  const int lane = tid & 31;
  const int m    = lane & 15;     // fragment row (M) / col (N) index
  const int h    = lane >> 4;     // half-wave select

  // ------------------------------------------------------------------
  // Phase 1: GroupNorm. Thread owns 64 contiguous channels (= 4 whole
  // groups of 16) of sample s -> stats are thread-local, no reductions.
  // ------------------------------------------------------------------
  {
    const int s    = tid >> 3;          // sample in block
    const int part = tid & 7;           // which 64-channel chunk
    const size_t gs = (size_t)blockIdx.x * SPB + s;
    const float* xrow = x + gs * Cdim + part * 64;
    float vals[64];
#pragma unroll
    for (int j = 0; j < 16; ++j)
      reinterpret_cast<float4*>(vals)[j] = reinterpret_cast<const float4*>(xrow)[j];
#pragma unroll
    for (int g = 0; g < 4; ++g) {
      float sum = 0.f, sq = 0.f;
#pragma unroll
      for (int j = 0; j < GS; ++j) { float v = vals[g * GS + j]; sum += v; sq += v * v; }
      const float mu   = sum * (1.f / GS);
      const float var  = sq * (1.f / GS) - mu * mu;
      const float rstd = rsqrtf(var + 1e-6f);
      const int grp = part * 4 + g;
      s_mu[s][grp] = mu;  s_rstd[s][grp] = rstd;
#pragma unroll
      for (int j = 0; j < GS; ++j) {
        const int c = part * 64 + g * GS + j;
        const float xn = (vals[g * GS + j] - mu) * rstd * gamma[c] + beta[c];
        s_xnb[s * XNB_STRIDE + c] = (__bf16)xn;
      }
    }
  }
  __syncthreads();

  // ------------------------------------------------------------------
  // Phase 2: k/q/v projections.  M=16 samples, N=128, K=512.
  // Wave w computes N-tiles {2w, 2w+1}; one A frag feeds 6 WMMAs.
  // ------------------------------------------------------------------
  {
    v8f acc[2][3] = {};
    for (int kt = 0; kt < 16; ++kt) {
      const int kb = kt * 32;
      const v16bf a = load_afrag_lds(s_xnb + m * XNB_STRIDE, kb, h);
#pragma unroll
      for (int nt = 0; nt < 2; ++nt) {
        const int ntg = wave * 2 + nt;
        v16bf bk, bq, bv;
        if (USE_BF) {
          const size_t fo = (size_t)(((ntg * 16 + kt) * 32 + lane)) * 16;
          bk = *(const v16bf*)(kfr + fo);
          bq = *(const v16bf*)(qfr + fo);
          bv = *(const v16bf*)(vfr + fo);
        } else {
          const int r = ntg * 16 + m;
          const size_t wo = (size_t)r * Cdim + kb + 16 * h;
          bk = load_bfrag_f32(kw + wo);
          bq = load_bfrag_f32(qw + wo);
          bv = load_bfrag_f32(vw + wo);
        }
        acc[nt][0] = wmma_bf16(a, bk, acc[nt][0]);
        acc[nt][1] = wmma_bf16(a, bq, acc[nt][1]);
        acc[nt][2] = wmma_bf16(a, bv, acc[nt][2]);
      }
    }
    // D layout: VGPR i -> sample = i + 8h ; N = lane&15
#pragma unroll
    for (int nt = 0; nt < 2; ++nt) {
      const int rc = (wave * 2 + nt) * 16 + m;
      const float bk = kbias[rc], bq = qbias[rc], bv = vbias[rc];
#pragma unroll
      for (int i = 0; i < 8; ++i) {
        const int sm = i + 8 * h;
        s_k[sm * KQV_STRIDE + rc] = acc[nt][0][i] + bk;
        *(float2*)&s_qv[sm * QV_STRIDE + 2 * rc] =
            make_float2(acc[nt][1][i] + bq, acc[nt][2][i] + bv);
      }
    }
  }
  __syncthreads();

  // ------------------------------------------------------------------
  // Phase 3: per-sample qmax/qmin (row-max of k_i*q is k_i*qmax or k_i*qmin)
  // ------------------------------------------------------------------
  if (tid < SPB) {
    float mx = -3.402823466e+38f, mn = 3.402823466e+38f;
    for (int j = 0; j < Rdim; ++j) {
      const float v = s_qv[tid * QV_STRIDE + 2 * j];
      mx = fmaxf(mx, v); mn = fminf(mn, v);
    }
    s_qmax[tid] = mx; s_qmin[tid] = mn;
  }
  __syncthreads();

  // ------------------------------------------------------------------
  // Phase 4: attention rows. out[s,i] = sum_j softmax_j(k_i*q_j) * v_j.
  // Iteration `sidx`: all 128 threads work on sample sidx, i = tid.
  // (q,v) interleaved -> one broadcast ds_load_b64 per j.
  // ------------------------------------------------------------------
  for (int sidx = 0; sidx < SPB; ++sidx) {
    const float ks   = s_k[sidx * KQV_STRIDE + tid];
    const float mrow = (ks >= 0.f) ? ks * s_qmax[sidx] : ks * s_qmin[sidx];
    float den = 0.f, num = 0.f;
    const float2* qv = (const float2*)(s_qv + sidx * QV_STRIDE);
#pragma unroll 8
    for (int j = 0; j < Rdim; ++j) {
      const float2 p = qv[j];
      const float e = __expf(ks * p.x - mrow);
      den += e;
      num += e * p.y;
    }
    // 1/sqrt(128)
    s_attn[sidx * ATT_STRIDE + tid] = (__bf16)((num / den) * 0.08838834764831843f);
  }
  __syncthreads();

  // ------------------------------------------------------------------
  // Phase 5: output projection M=16, N=512, K=128 + fused epilogue:
  //   out = xn (exact fp32, recomputed from mu/rstd) + GEMM + r_b
  // Wave w handles N-tiles 8w..8w+7; A frags hoisted across N-tiles.
  // ------------------------------------------------------------------
  {
    v16bf afr[4];
#pragma unroll
    for (int kt = 0; kt < 4; ++kt)
      afr[kt] = load_afrag_lds(s_attn + m * ATT_STRIDE, kt * 32, h);

    for (int nt = 0; nt < 8; ++nt) {
      const int ntg = wave * 8 + nt;
      const int c = ntg * 16 + m;     // lane's output channel (N)
      v8f acc = {};
#pragma unroll
      for (int kt = 0; kt < 4; ++kt) {
        v16bf b;
        if (USE_BF) {
          const size_t fo = (size_t)(((ntg * 4 + kt) * 32 + lane)) * 16;
          b = *(const v16bf*)(rfr + fo);
        } else {
          // B[k=r][n=c] = rw[c*R + r]; 16 contiguous r per lane
          b = load_bfrag_f32(rw + (size_t)c * Rdim + kt * 32 + 16 * h);
        }
        acc = wmma_bf16(afr[kt], b, acc);
      }
      const float rb  = rbias[c];
      const float gmc = gamma[c];
      const float btc = beta[c];
      const int   grp = c >> 4;
#pragma unroll
      for (int i = 0; i < 8; ++i) {
        const int sm = i + 8 * h;
        const size_t gs = (size_t)blockIdx.x * SPB + sm;
        const float xv = x[gs * Cdim + c];
        const float xn = (xv - s_mu[sm][grp]) * s_rstd[sm][grp] * gmc + btc;
        out[gs * Cdim + c] = xn + acc[i] + rb;
      }
    }
  }
}

extern "C" void kernel_launch(void* const* d_in, const int* in_sizes, int n_in,
                              void* d_out, int out_size, void* d_ws, size_t ws_size,
                              hipStream_t stream) {
  const float* x     = (const float*)d_in[0];
  const float* gamma = (const float*)d_in[1];
  const float* beta  = (const float*)d_in[2];
  const float* kw    = (const float*)d_in[3];
  const float* kb    = (const float*)d_in[4];
  const float* qw    = (const float*)d_in[5];
  const float* qb    = (const float*)d_in[6];
  const float* vw    = (const float*)d_in[7];
  const float* vb    = (const float*)d_in[8];
  const float* rw    = (const float*)d_in[9];
  const float* rb    = (const float*)d_in[10];
  float* out = (float*)d_out;

  const int nB   = in_sizes[0] / Cdim;   // 16384
  const int grid = nB / SPB;             // 1024 blocks of 16 samples

  const bool use_bf = ws_size >= (size_t)WS_TOTAL_ELEMS * sizeof(unsigned short);
  if (use_bf) {
    unsigned short* wsb = (unsigned short*)d_ws;
    prep_weights<<<WS_TOTAL_ELEMS / 256, 256, 0, stream>>>(kw, qw, vw, rw, wsb);
    attn_block_fused<true><<<grid, NTHR, 0, stream>>>(x, gamma, beta,
        kw, kb, qw, qb, vw, vb, rw, rb, wsb, out);
  } else {
    attn_block_fused<false><<<grid, NTHR, 0, stream>>>(x, gamma, beta,
        kw, kb, qw, qb, vw, vb, rw, rb, (const unsigned short*)nullptr, out);
  }
}
